// FlashAttentionVarlen_56762287784564
// MI455X (gfx1250) — matrix operand order repaired and checked by reading, not compiled, not run on hardware
//
#include <hip/hip_runtime.h>

typedef __attribute__((ext_vector_type(16))) _Float16 v16h;
typedef __attribute__((ext_vector_type(8)))  _Float16 v8h;
typedef __attribute__((ext_vector_type(8)))  float    v8f;
typedef __attribute__((ext_vector_type(4)))  int      v4i;

#define NEG_BIG (-1e30f)

constexpr int HEADS = 16;
constexpr int DHEAD = 128;
constexpr int BM    = 64;    // Q rows per workgroup (16 per wave)
constexpr int BN    = 64;    // K rows per tile
constexpr int KSTR  = 144;   // sK row stride in halves (288B: 16B/32B aligned rows)
constexpr int VTSTR = 80;    // sVt row stride in halves (160B: 16B/32B aligned rows)

#if defined(__HIP_DEVICE_COMPILE__) && \
    __has_builtin(__builtin_amdgcn_global_load_async_to_lds_b128) && \
    __has_builtin(__builtin_amdgcn_s_wait_asynccnt)
#define HAS_ASYNC 1
#else
#define HAS_ASYNC 0
#endif

// builtin signature: (v4i addrspace(1)*, v4i addrspace(3)*, imm offset, imm cpol)
#define AS1_V4I(p) ((__attribute__((address_space(1))) v4i*)(p))
#define AS3_V4I(p) ((__attribute__((address_space(3))) v4i*)(p))

__device__ __forceinline__ v16h ldv16(const _Float16* p) {
  return *reinterpret_cast<const v16h*>(p);
}

// ---- shared per-K-tile compute: 16 WMMA scores + online softmax + 16 WMMA PV ----
__device__ __forceinline__ void fa_tile_compute(
    const v16h aq[4], const _Float16* __restrict__ sKt,
    const _Float16* __restrict__ sVtt, _Float16* __restrict__ sPw,
    int kb, int k_len, int lr, int hi, int koff8, int koff16,
    float* mrow, float* lrow, v8f* o) {
  // S = Q @ K^T : four 16x16 f32 accumulators
  v8f sa[4];
#pragma unroll
  for (int nt = 0; nt < 4; ++nt) {
    v8f acc = {};
#pragma unroll
    for (int c = 0; c < 4; ++c) {
      v16h bk = ldv16(sKt + (nt * 16 + lr) * KSTR + c * 32 + koff16);
      acc = __builtin_amdgcn_wmma_f32_16x16x32_f16(false, aq[c], false, bk,
                                                   (short)0, acc, false, false);
    }
    if (kb + nt * 16 + lr >= k_len) {  // ragged-K column mask
#pragma unroll
      for (int j = 0; j < 8; ++j) acc[j] = NEG_BIG;
    }
    sa[nt] = acc;
  }

  // online softmax (row stats via half-wave xor shuffles)
  float alpha[8];
#pragma unroll
  for (int j = 0; j < 8; ++j) {
    float t = fmaxf(fmaxf(sa[0][j], sa[1][j]), fmaxf(sa[2][j], sa[3][j]));
#pragma unroll
    for (int xm = 8; xm >= 1; xm >>= 1)
      t = fmaxf(t, __shfl_xor(t, xm, 32));
    float mn = fmaxf(mrow[j], t);
    alpha[j] = __expf(mrow[j] - mn);
    mrow[j]  = mn;
  }
  float tsum[8];
#pragma unroll
  for (int j = 0; j < 8; ++j) tsum[j] = 0.f;
#pragma unroll
  for (int nt = 0; nt < 4; ++nt) {
#pragma unroll
    for (int j = 0; j < 8; ++j) {
      float p = __expf(sa[nt][j] - mrow[j]);
      tsum[j] += p;
      sPw[(hi * 8 + j) * 64 + nt * 16 + lr] = (_Float16)p;  // C-layout -> LDS
    }
  }
#pragma unroll
  for (int j = 0; j < 8; ++j) {
#pragma unroll
    for (int xm = 8; xm >= 1; xm >>= 1)
      tsum[j] += __shfl_xor(tsum[j], xm, 32);
    lrow[j] = lrow[j] * alpha[j] + tsum[j];
  }
#pragma unroll
  for (int dt = 0; dt < 8; ++dt)
#pragma unroll
    for (int j = 0; j < 8; ++j) o[dt][j] *= alpha[j];

  // O += P @ V : read P back in A layout, V^T tiles as B
#pragma unroll
  for (int kk = 0; kk < BN; kk += 32) {
    v16h pa = {};
#pragma unroll
    for (int i = 0; i < 8; ++i) {
      pa[i]     = sPw[lr * 64 + kk + koff16 + i];
      pa[8 + i] = sPw[lr * 64 + kk + koff16 + 8 + i];
    }
#pragma unroll
    for (int dt = 0; dt < 8; ++dt) {
      v16h bv = ldv16(sVtt + (dt * 16 + lr) * VTSTR + kk + koff16);
      o[dt] = __builtin_amdgcn_wmma_f32_16x16x32_f16(false, pa, false, bv,
                                                     (short)0, o[dt], false, false);
    }
  }
}

// ================= preprocessing: f32 -> f16 =================
__global__ void cvt_f32_f16(const float* __restrict__ src,
                            _Float16* __restrict__ dst, long long n) {
  long long i = ((long long)blockIdx.x * blockDim.x + threadIdx.x) * 8;
  if (i + 8 > n) return;
  float4 a = *reinterpret_cast<const float4*>(src + i);
  float4 b = *reinterpret_cast<const float4*>(src + i + 4);
  v8h o;
  o[0] = (_Float16)a.x; o[1] = (_Float16)a.y; o[2] = (_Float16)a.z; o[3] = (_Float16)a.w;
  o[4] = (_Float16)b.x; o[5] = (_Float16)b.y; o[6] = (_Float16)b.z; o[7] = (_Float16)b.w;
  *reinterpret_cast<v8h*>(dst + i) = o;
}

// V [t][h][d] f32 -> Vt [b][h][d][s] f16 (LDS-tiled transpose, 64s x 128d per block)
__global__ void cvt_v_transpose(const float* __restrict__ v,
                                const int* __restrict__ cuk,
                                _Float16* __restrict__ vt, int S) {
  __shared__ alignas(32) _Float16 tile[DHEAD * 72];  // [d][s], stride 72 (144B rows)
  const int s0 = blockIdx.x * 64, b = blockIdx.y, h = blockIdx.z;
  const int k_start = cuk[b];
  const int len     = cuk[b + 1] - k_start;
  const int tid = threadIdx.x;  // 256 threads
  for (int it = tid; it < 64 * 32; it += 256) {
    int r = it >> 5, c4 = (it & 31) << 2;
    float4 f = make_float4(0.f, 0.f, 0.f, 0.f);
    if (s0 + r < len)
      f = *reinterpret_cast<const float4*>(
          v + (size_t)(k_start + s0 + r) * (HEADS * DHEAD) + h * DHEAD + c4);
    tile[(c4 + 0) * 72 + r] = (_Float16)f.x;
    tile[(c4 + 1) * 72 + r] = (_Float16)f.y;
    tile[(c4 + 2) * 72 + r] = (_Float16)f.z;
    tile[(c4 + 3) * 72 + r] = (_Float16)f.w;
  }
  __syncthreads();
  _Float16* outb = vt + ((size_t)(b * HEADS + h) * DHEAD) * S + s0;
  for (int it = tid; it < DHEAD * 8; it += 256) {
    int d = it >> 3, c = it & 7;
    *reinterpret_cast<v8h*>(outb + (size_t)d * S + c * 8) =
        *reinterpret_cast<const v8h*>(tile + d * 72 + c * 8);
  }
}

// ================= main kernel: async double-buffered f16 path =================
__launch_bounds__(128)
__global__ void fa_varlen_wmma_async(const float* __restrict__ q,
                                     const _Float16* __restrict__ kh,
                                     const _Float16* __restrict__ vt,
                                     const int* __restrict__ cuq,
                                     const int* __restrict__ cuk,
                                     float* __restrict__ out, int S) {
  __shared__ alignas(32) _Float16 sKb[2][BN * KSTR];
  __shared__ alignas(32) _Float16 sVb[2][DHEAD * VTSTR];
  __shared__ alignas(32) _Float16 sP[4 * 16 * 64];

  const int qt = blockIdx.x, b = blockIdx.y, h = blockIdx.z;
  const int q_start = cuq[b];
  const int q_len   = cuq[b + 1] - q_start;
  const int k_start = cuk[b];
  const int k_len   = cuk[b + 1] - k_start;
  const int q0 = qt * BM;
  if (q0 >= q_len) return;

  const int tid = threadIdx.x;
  const int lane = tid & 31, wv = tid >> 5;
  const int lr = lane & 15, hi = lane >> 4;
  const int koff8 = hi * 8, koff16 = hi * 16;
  const float scale = 0.08838834764831845f;  // 1/sqrt(128)

  const _Float16* khb = kh + ((size_t)k_start * HEADS + h) * DHEAD;
  const _Float16* vtb = vt + ((size_t)(b * HEADS + h) * DHEAD) * S;

  // Q rows -> A fragments (f32 global, once per block; pre-scaled)
  v16h aq[4];
  {
    int qr = q0 + wv * 16 + lr;
    bool ok = qr < q_len;
    int qrc = ok ? qr : (q_len - 1);
    const float* qb = q + (size_t)(q_start + qrc) * (HEADS * DHEAD) + h * DHEAD;
#pragma unroll
    for (int c = 0; c < 4; ++c)
#pragma unroll
      for (int i = 0; i < 8; ++i) {
        aq[c][i]     = (_Float16)(ok ? qb[c * 32 + koff16 + i] * scale : 0.f);
        aq[c][8 + i] = (_Float16)(ok ? qb[c * 32 + koff16 + 8 + i] * scale : 0.f);
      }
  }

  float mrow[8], lrow[8];
  v8f o[8];
#pragma unroll
  for (int j = 0; j < 8; ++j) { mrow[j] = NEG_BIG; lrow[j] = 0.f; }
#pragma unroll
  for (int dt = 0; dt < 8; ++dt) o[dt] = {};
  _Float16* sPw = sP + wv * (16 * 64);

  // issue one K+Vt tile copy (16 b128 async VMEM ops per wave)
  auto stage = [&](int buf, int kb) {
#pragma unroll
    for (int i = 0; i < 8; ++i) {
      int id = tid + i * 128;
      {  // K tile: 64 rows x 16 chunks of 16B
        int r = id >> 4, c = id & 15;
        int rc = (kb + r < k_len) ? kb + r : k_len - 1;
        const _Float16* gp = khb + (size_t)rc * (HEADS * DHEAD) + c * 8;
        _Float16* lp = &sKb[buf][r * KSTR + c * 8];
#if HAS_ASYNC
        __builtin_amdgcn_global_load_async_to_lds_b128(AS1_V4I(gp), AS3_V4I(lp), 0, 0);
#else
        *reinterpret_cast<v8h*>(lp) = *reinterpret_cast<const v8h*>(gp);
#endif
      }
      {  // Vt tile: 128 d-rows x 8 chunks of 16B
        int d = id >> 3, c = id & 7;
        const _Float16* gp = vtb + (size_t)d * S + kb + c * 8;
        _Float16* lp = &sVb[buf][d * VTSTR + c * 8];
#if HAS_ASYNC
        __builtin_amdgcn_global_load_async_to_lds_b128(AS1_V4I(gp), AS3_V4I(lp), 0, 0);
#else
        *reinterpret_cast<v8h*>(lp) = *reinterpret_cast<const v8h*>(gp);
#endif
      }
    }
  };

  const int ntiles = (k_len + BN - 1) / BN;
  if (ntiles > 0) stage(0, 0);
  for (int it = 0; it < ntiles; ++it) {
    const int kb = it * BN;
    const bool more = (it + 1) < ntiles;
    if (more) stage((it + 1) & 1, kb + BN);  // overlap next tile with this compute
#if HAS_ASYNC
    if (more) __builtin_amdgcn_s_wait_asynccnt(16);  // tile `it` done (FIFO)
    else      __builtin_amdgcn_s_wait_asynccnt(0);
#endif
    __syncthreads();
    fa_tile_compute(aq, sKb[it & 1], sVb[it & 1], sPw, kb, k_len,
                    lr, hi, koff8, koff16, mrow, lrow, o);
    __syncthreads();
  }

  float rinv[8];
#pragma unroll
  for (int j = 0; j < 8; ++j) rinv[j] = (lrow[j] > 0.f) ? 1.f / lrow[j] : 0.f;
#pragma unroll
  for (int j = 0; j < 8; ++j) {
    int ro = q0 + wv * 16 + hi * 8 + j;
    if (ro < q_len) {
      float* ob = out + (size_t)(q_start + ro) * (HEADS * DHEAD) + h * DHEAD + lr;
#pragma unroll
      for (int dt = 0; dt < 8; ++dt) ob[dt * 16] = o[dt][j] * rinv[j];
    }
  }
}

// ============== fallback: all-f32 synchronous staging (round-1 kernel) ==============
__launch_bounds__(128)
__global__ void fa_varlen_wmma(const float* __restrict__ q,
                               const float* __restrict__ k,
                               const float* __restrict__ v,
                               const int* __restrict__ cuq,
                               const int* __restrict__ cuk,
                               float* __restrict__ out) {
  __shared__ alignas(32) _Float16 sK[BN * KSTR];
  __shared__ alignas(32) _Float16 sVt[DHEAD * VTSTR];
  __shared__ alignas(32) _Float16 sP[4 * 16 * 64];

  const int qt = blockIdx.x, b = blockIdx.y, h = blockIdx.z;
  const int q_start = cuq[b];
  const int q_len   = cuq[b + 1] - q_start;
  const int k_start = cuk[b];
  const int k_len   = cuk[b + 1] - k_start;
  const int q0 = qt * BM;
  if (q0 >= q_len) return;

  const int tid = threadIdx.x;
  const int lane = tid & 31, wv = tid >> 5;
  const int lr = lane & 15, hi = lane >> 4;
  const int koff8 = hi * 8, koff16 = hi * 16;
  const float scale = 0.08838834764831845f;

  v16h aq[4];
  {
    int qr = q0 + wv * 16 + lr;
    bool ok = qr < q_len;
    int qrc = ok ? qr : (q_len - 1);
    const float* qb = q + (size_t)(q_start + qrc) * (HEADS * DHEAD) + h * DHEAD;
#pragma unroll
    for (int c = 0; c < 4; ++c)
#pragma unroll
      for (int i = 0; i < 8; ++i) {
        aq[c][i]     = (_Float16)(ok ? qb[c * 32 + koff16 + i] * scale : 0.f);
        aq[c][8 + i] = (_Float16)(ok ? qb[c * 32 + koff16 + 8 + i] * scale : 0.f);
      }
  }

  float mrow[8], lrow[8];
  v8f o[8];
#pragma unroll
  for (int j = 0; j < 8; ++j) { mrow[j] = NEG_BIG; lrow[j] = 0.f; }
#pragma unroll
  for (int dt = 0; dt < 8; ++dt) o[dt] = {};
  _Float16* sPw = sP + wv * (16 * 64);

  for (int kb = 0; kb < k_len; kb += BN) {
    const int nrows = (k_len - kb < BN) ? (k_len - kb) : BN;
#pragma unroll 4
    for (int it = tid; it < BN * 32; it += 128) {
      int r = it >> 5, c4 = (it & 31) << 2;
      float4 fk = make_float4(0.f, 0.f, 0.f, 0.f);
      float4 fv = make_float4(0.f, 0.f, 0.f, 0.f);
      if (r < nrows) {
        size_t off = (size_t)(k_start + kb + r) * (HEADS * DHEAD) + h * DHEAD + c4;
        fk = *reinterpret_cast<const float4*>(k + off);
        fv = *reinterpret_cast<const float4*>(v + off);
        if (kb + BN < k_len) {
          __builtin_prefetch(k + off + (size_t)BN * HEADS * DHEAD, 0, 1);
          __builtin_prefetch(v + off + (size_t)BN * HEADS * DHEAD, 0, 1);
        }
      }
      _Float16* kd = sK + r * KSTR + c4;
      kd[0] = (_Float16)fk.x; kd[1] = (_Float16)fk.y;
      kd[2] = (_Float16)fk.z; kd[3] = (_Float16)fk.w;
      sVt[(c4 + 0) * VTSTR + r] = (_Float16)fv.x;
      sVt[(c4 + 1) * VTSTR + r] = (_Float16)fv.y;
      sVt[(c4 + 2) * VTSTR + r] = (_Float16)fv.z;
      sVt[(c4 + 3) * VTSTR + r] = (_Float16)fv.w;
    }
    __syncthreads();
    fa_tile_compute(aq, sK, sVt, sPw, kb, k_len, lr, hi, koff8, koff16,
                    mrow, lrow, o);
    __syncthreads();
  }

  float rinv[8];
#pragma unroll
  for (int j = 0; j < 8; ++j) rinv[j] = (lrow[j] > 0.f) ? 1.f / lrow[j] : 0.f;
#pragma unroll
  for (int j = 0; j < 8; ++j) {
    int ro = q0 + wv * 16 + hi * 8 + j;
    if (ro < q_len) {
      float* ob = out + (size_t)(q_start + ro) * (HEADS * DHEAD) + h * DHEAD + lr;
#pragma unroll
      for (int dt = 0; dt < 8; ++dt) ob[dt * 16] = o[dt][j] * rinv[j];
    }
  }
}

extern "C" void kernel_launch(void* const* d_in, const int* in_sizes, int n_in,
                              void* d_out, int out_size, void* d_ws, size_t ws_size,
                              hipStream_t stream) {
  (void)n_in; (void)out_size;
  const float* q   = (const float*)d_in[0];
  const float* k   = (const float*)d_in[1];
  const float* v   = (const float*)d_in[2];
  const int*   cuq = (const int*)d_in[3];
  const int*   cuk = (const int*)d_in[4];
  float*       out = (float*)d_out;

  const long long elems = (long long)in_sizes[0];           // T*H*D
  const int T = (int)(elems / (HEADS * DHEAD));
  const int B = in_sizes[3] - 1;
  if (B <= 0 || T <= 0) return;
  const bool uniform = (T % B) == 0;
  const int S = uniform ? T / B : 0;
  const int qtiles = (S > 0) ? (S + BM - 1) / BM : 0;
  const size_t need = (size_t)elems * 4;  // kh (2B) + vt (2B)

  if (uniform && S > 0 && (S % BN) == 0 && ws_size >= need && d_ws) {
    _Float16* kh = (_Float16*)d_ws;
    _Float16* vt = kh + elems;
    const int cvb = (int)((elems / 8 + 255) / 256);
    cvt_f32_f16<<<cvb, 256, 0, stream>>>(k, kh, elems);
    cvt_v_transpose<<<dim3(S / 64, B, HEADS), 256, 0, stream>>>(v, cuk, vt, S);
    dim3 grid(qtiles, B, HEADS);
    fa_varlen_wmma_async<<<grid, 128, 0, stream>>>(q, kh, vt, cuq, cuk, out, S);
  } else {
    const int qt2 = ((T / (B > 0 ? B : 1)) + BM - 1) / BM;
    dim3 grid(qt2 > 0 ? qt2 : 1, B, HEADS);
    fa_varlen_wmma<<<grid, 128, 0, stream>>>(q, k, v, cuq, cuk, out);
  }
}
